// STDLoss_75222057222180
// MI455X (gfx1250) — compile-verified, matrix operand order checked by reading, not run on hardware
//
#include <hip/hip_runtime.h>
#include <hip/hip_bf16.h>
#include <math.h>

typedef __attribute__((ext_vector_type(16))) _Float16 v16h;
typedef __attribute__((ext_vector_type(8)))  float    v8f;

#define C_DIM 64
#define S_DIM 8
#define W_DIM 1024
#define K_DIM 128

// One workgroup (256 threads = 8 wave32) per w-slice.
// Builds n[c][k] histogram in LDS, then:
//  - in-class partial: sum over (c,k) of sqrt(n*(8-n))            (scaled later)
//  - inter-class partial: sum over k of std_C(n/S) via WMMA column sums
__global__ __launch_bounds__(256) void std_loss_block(
    const int* __restrict__ target,
    float* __restrict__ wsInter,
    float* __restrict__ wsIn) {
  __shared__ unsigned int cnt[C_DIM * K_DIM];   // 8192 u32 = 32 KB
  __shared__ float redA[8];
  __shared__ float redB[8];

  const int t = threadIdx.x;
  const int w = blockIdx.x;

  // Zero the histogram.
  #pragma unroll
  for (int i = 0; i < 32; ++i) cnt[t + 256 * i] = 0u;
  __syncthreads();

  // Histogram over shots: linear cs = c*S + s, address (c*S+s)*W + w.
  #pragma unroll
  for (int r = 0; r < 2; ++r) {
    const int cs = t + 256 * r;                       // [0, 512)
    const int k = target[(size_t)cs * W_DIM + w] - 1; // [0, 128)
    atomicAdd(&cnt[(cs >> 3) * K_DIM + k], 1u);
  }
  __syncthreads();

  // ---- in-class term: sum_{c,k} sqrt(n*(S-n)) ----
  float pin = 0.0f;
  #pragma unroll
  for (int i = 0; i < 32; ++i) {
    const unsigned int n = cnt[t + 256 * i];
    pin += sqrtf((float)(n * (S_DIM - n)));
  }

  // ---- inter-class term via WMMA ----
  // wave -> 16-wide k tile; lane n (and n+16) own column kbase + (lane&15).
  // B operand as filled here: lane n = col n, K=0..15 in its 16
  // halves; lane n+16 = col n, K=16..31. With A == all-ones, any K ordering
  // inside a column yields the same column sum, so we just need the right
  // lane <-> column mapping.
  const int wave  = t >> 5;          // 0..7
  const int lane  = t & 31;
  const int k     = (wave << 4) + (lane & 15);
  const int chalf = lane & 16;       // which 16 classes of a 32-class chunk

  v16h ones;
  #pragma unroll
  for (int j = 0; j < 16; ++j) ones[j] = (_Float16)1.0f;

  v16h bn0, bn1, bq0, bq1;
  #pragma unroll
  for (int j = 0; j < 16; ++j) {
    const unsigned int n0 = cnt[(chalf + j) * K_DIM + k];       // classes 0..31
    const unsigned int n1 = cnt[(32 + chalf + j) * K_DIM + k];  // classes 32..63
    bn0[j] = (_Float16)(float)n0;            // n   (<= 8, exact in f16)
    bn1[j] = (_Float16)(float)n1;
    bq0[j] = (_Float16)(float)(n0 * n0);     // n^2 (<= 64, exact in f16)
    bq1[j] = (_Float16)(float)(n1 * n1);
  }

  v8f acc1 = {};  // S1[k] = sum_c n
  v8f acc2 = {};  // S2[k] = sum_c n^2
  acc1 = __builtin_amdgcn_wmma_f32_16x16x32_f16(false, ones, false, bn0,
                                                (short)0, acc1, false, false);
  acc1 = __builtin_amdgcn_wmma_f32_16x16x32_f16(false, ones, false, bn1,
                                                (short)0, acc1, false, false);
  acc2 = __builtin_amdgcn_wmma_f32_16x16x32_f16(false, ones, false, bq0,
                                                (short)0, acc2, false, false);
  acc2 = __builtin_amdgcn_wmma_f32_16x16x32_f16(false, ones, false, bq1,
                                                (short)0, acc2, false, false);

  // D row 0 (VGPR0): lanes 0-15 give column sums for cols 0-15; lanes 16-31
  // carry row M=8 which (A all-ones) equals the same column sums -> mask them.
  const float S1 = acc1[0];
  const float S2 = acc2[0];
  const float var_n = (S2 - S1 * S1 * (1.0f / (float)C_DIM)) *
                      (1.0f / (float)(C_DIM - 1));
  const float stdv = sqrtf(fmaxf(var_n, 0.0f)) * (1.0f / (float)S_DIM);
  float pinter = (lane < 16) ? stdv : 0.0f;

  // Wave32 butterfly reduction, then cross-wave via LDS (fixed order).
  #pragma unroll
  for (int o = 16; o > 0; o >>= 1) {
    pin    += __shfl_xor(pin, o, 32);
    pinter += __shfl_xor(pinter, o, 32);
  }
  if (lane == 0) { redA[wave] = pin; redB[wave] = pinter; }
  __syncthreads();
  if (t == 0) {
    float a = 0.0f, b = 0.0f;
    #pragma unroll
    for (int i = 0; i < 8; ++i) { a += redA[i]; b += redB[i]; }
    wsIn[w]    = a;   // unscaled sum of sqrt(n*(S-n)) for this w
    wsInter[w] = b;   // sum over k of std_C for this w
  }
}

// Deterministic final reduction over the 1024 per-w partials.
__global__ __launch_bounds__(256) void std_loss_final(
    const float* __restrict__ wsInter,
    const float* __restrict__ wsIn,
    float* __restrict__ out) {
  __shared__ float ra[256];
  __shared__ float rb[256];
  const int t = threadIdx.x;
  float a = 0.0f, b = 0.0f;
  for (int i = t; i < W_DIM; i += 256) { a += wsInter[i]; b += wsIn[i]; }
  ra[t] = a; rb[t] = b;
  __syncthreads();
  for (int s = 128; s > 0; s >>= 1) {
    if (t < s) { ra[t] += ra[t + s]; rb[t] += rb[t + s]; }
    __syncthreads();
  }
  if (t == 0) {
    // in_class scale: 1 / (C * sqrt(S*(S-1))) = 1 / (64 * sqrt(56))
    const float inScale = 1.0f / (64.0f * 7.4833147735478827f);
    out[0] = 1.0f / ra[0] + rb[0] * inScale;
  }
}

extern "C" void kernel_launch(void* const* d_in, const int* in_sizes, int n_in,
                              void* d_out, int out_size, void* d_ws, size_t ws_size,
                              hipStream_t stream) {
  (void)in_sizes; (void)n_in; (void)out_size; (void)ws_size;
  const int* target = (const int*)d_in[0];
  float* ws = (float*)d_ws;
  float* wsInter = ws;            // 1024 floats
  float* wsIn    = ws + W_DIM;    // 1024 floats

  std_loss_block<<<W_DIM, 256, 0, stream>>>(target, wsInter, wsIn);
  std_loss_final<<<1, 256, 0, stream>>>(wsInter, wsIn, (float*)d_out);
}
